// FastAttention_6262062318048
// MI455X (gfx1250) — matrix operand order repaired and checked by reading, not compiled, not run on hardware
//
#include <hip/hip_runtime.h>
#include <hip/hip_bf16.h>
#include <cmath>

// ---------------------------------------------------------------------------
// Performer (FAVOR+) attention for MI455X / gfx1250, wave32, f16 WMMA.
// B=4 H=16 N=4096 D=E=64 M=256, fp32 in/out.
// Fused streaming design: q'/k' feature maps never touch HBM; workspace holds
// only ctx^T [BH,64,256] f32 (4MB) + ksum + max partials (~4.1MB).
// LDS fragment traffic vectorized to ds_load_b128/ds_store_b128, except scalar v B-fragment loads.
// ---------------------------------------------------------------------------

#define BDIM   4
#define HDIM   16
#define BH     (BDIM * HDIM)        // 64
#define NSEQ   4096
#define DDIM   64
#define MFEA   256
#define EDIM   64

#define NORMALIZER 0.35355339059327373f   // 64^-0.25 (folded into projection)
#define DIAGC      0.0625f                // 0.5 * 64^-0.5
#define RATIO      0.0625f                // 256^-0.5
#define FEPS       1e-4f
#define NEG_INF    (-3.4e38f)

typedef _Float16 v16h __attribute__((ext_vector_type(16)));
typedef _Float16 v8h  __attribute__((ext_vector_type(8)));
typedef float    v8f  __attribute__((ext_vector_type(8)));

// ---- WMMA fragment helpers (contiguous k = 16*(lane/16) + s, wave32) ----------

// A fragment, 16x32 f16 (MxK), from row-major [m][k] LDS tile.
// Lane l<16: M=l, K in {0..15}; lane>=16: M=l-16, K in {16..31}.
// Each lane = two contiguous 8-half runs -> 2x ds_load_b128.
__device__ __forceinline__ v16h ld_a16(const _Float16* base, int stride,
                                       int lane, int kbase) {
  const int m  = lane & 15;
  const int kh = (lane < 16) ? 0 : 16;
  const _Float16* p = base + m * stride + kbase + kh;
  const v8h lo = *(const v8h*)p;
  const v8h hi = *(const v8h*)(p + 8);
  return __builtin_shufflevector(lo, hi, 0, 1, 2, 3, 4, 5, 6, 7,
                                 8, 9, 10, 11, 12, 13, 14, 15);
}

// B fragment, 32x16 f16 (KxN), from per-lane fragment layout [lane][16].
__device__ __forceinline__ v16h ld_bfrag(const _Float16* tile, int lane) {
  const _Float16* p = tile + lane * 16;
  const v8h lo = *(const v8h*)p;
  const v8h hi = *(const v8h*)(p + 8);
  return __builtin_shufflevector(lo, hi, 0, 1, 2, 3, 4, 5, 6, 7,
                                 8, 9, 10, 11, 12, 13, 14, 15);
}

// B fragment from K-contiguous (transposed) storage [n][k], stride in halves.
// Lane l<16: K=kbase..+15; lane>=16: K=kbase+16..+31; col N = lane&15.
__device__ __forceinline__ v16h ld_bT(const _Float16* base, int stride,
                                      int lane, int kbase) {
  const _Float16* p = base + (lane & 15) * stride + kbase +
                      ((lane < 16) ? 0 : 16);
  const v8h lo = *(const v8h*)p;
  const v8h hi = *(const v8h*)(p + 8);
  return __builtin_shufflevector(lo, hi, 0, 1, 2, 3, 4, 5, 6, 7,
                                 8, 9, 10, 11, 12, 13, 14, 15);
}

// B fragment, scalar fallback from row-major [k][n] tile (v rows only).
__device__ __forceinline__ v16h ld_b16s(const _Float16* base, int stride,
                                        int lane, int kbase) {
  v16h b;
  const int n  = lane & 15;
  const int kh = (lane < 16) ? 0 : 16;
#pragma unroll
  for (int e = 0; e < 16; ++e) b[e] = base[(kbase + kh + e) * stride + n];
  return b;
}

__device__ __forceinline__ v8f wmma_f16(v16h a, v16h b, v8f c) {
  return __builtin_amdgcn_wmma_f32_16x16x32_f16(false, a, false, b,
                                                (short)0, c, false, false);
}

__device__ __forceinline__ v8f zero8() {
  v8f z = {0.f, 0.f, 0.f, 0.f, 0.f, 0.f, 0.f, 0.f};
  return z;
}

// Fill projection into B-fragment layout: 32 tiles (16 mt x 2 ksteps) of
// [32 lanes][16 halves]. Thread t owns projection row m=t; contiguous d maps
// to contiguous elem -> vectorized b128 stores.
__device__ __forceinline__ void fill_pfrag(_Float16* p_lds,
                                           const float* __restrict__ proj,
                                           int tid) {
  const int m = tid;
  const float* src = proj + m * DDIM;
  const int mt = m >> 4, N = m & 15;
#pragma unroll
  for (int g = 0; g < 4; ++g) {            // four 16-d subgroups
    const int kstep = g >> 1;
    const int lslot = N + ((g & 1) ? 16 : 0);
    _Float16* dst = p_lds + ((mt * 2 + kstep) * 32 + lslot) * 16;
    v8h h0, h1;
#pragma unroll
    for (int j = 0; j < 8; ++j) {
      h0[j] = (_Float16)(src[g * 16 + j] * NORMALIZER);
      h1[j] = (_Float16)(src[g * 16 + 8 + j] * NORMALIZER);
    }
    *(v8h*)dst = h0;
    *((v8h*)dst + 1) = h1;
  }
}

// ---------------------------------------------------------------------------
// Kernel Z: zero ctx^T + ksum accumulators in workspace.
// ---------------------------------------------------------------------------
__global__ __launch_bounds__(256) void perf_zero(float* __restrict__ p, int n) {
  for (int i = blockIdx.x * blockDim.x + threadIdx.x; i < n;
       i += gridDim.x * blockDim.x)
    p[i] = 0.f;
}

// ---------------------------------------------------------------------------
// Kernel A: per-block max of k·Pᵀ. 128 rows per block. grid = BH*32.
// ---------------------------------------------------------------------------
__global__ __launch_bounds__(256) void perf_kmax(const float* __restrict__ k,
                                                 const float* __restrict__ proj,
                                                 float* __restrict__ partial) {
  __shared__ alignas(16) _Float16 p_lds[DDIM * MFEA];   // B-fragment layout
  __shared__ alignas(16) _Float16 a_lds[128 * DDIM];    // k rows row-major
  __shared__ float wred[8];

  const int tid = threadIdx.x, lane = tid & 31, w = tid >> 5;
  const int bh = blockIdx.x >> 5, blk = blockIdx.x & 31;

  fill_pfrag(p_lds, proj, tid);
  {  // 32 contiguous halves per thread, packed b128 stores
    const float* s = k + ((size_t)bh * NSEQ + (size_t)blk * 128) * DDIM +
                     tid * 32;
    _Float16* dst = a_lds + tid * 32;
#pragma unroll
    for (int g = 0; g < 4; ++g) {
      v8h h;
#pragma unroll
      for (int j = 0; j < 8; ++j) h[j] = (_Float16)s[g * 8 + j];
      *(v8h*)(dst + g * 8) = h;
    }
  }
  __syncthreads();

  const v16h a0 = ld_a16(a_lds + w * 16 * DDIM, DDIM, lane, 0);
  const v16h a1 = ld_a16(a_lds + w * 16 * DDIM, DDIM, lane, 32);
  float lmax = NEG_INF;
  for (int mt = 0; mt < 16; ++mt) {
    v8f acc = zero8();
    acc = wmma_f16(a0, ld_bfrag(p_lds + (mt * 2 + 0) * 512, lane), acc);
    acc = wmma_f16(a1, ld_bfrag(p_lds + (mt * 2 + 1) * 512, lane), acc);
#pragma unroll
    for (int e = 0; e < 8; ++e) lmax = fmaxf(lmax, acc[e]);
  }
#pragma unroll
  for (int off = 1; off < 32; off <<= 1)
    lmax = fmaxf(lmax, __shfl_xor(lmax, off, 32));
  if (lane == 0) wred[w] = lmax;
  __syncthreads();
  if (tid == 0) {
    float m = wred[0];
#pragma unroll
    for (int i = 1; i < 8; ++i) m = fmaxf(m, wred[i]);
    partial[blockIdx.x] = m;
  }
}

// ---------------------------------------------------------------------------
// Kernel B: reduce partial maxes -> global scalar. 1 block.
// ---------------------------------------------------------------------------
__global__ __launch_bounds__(256) void perf_maxreduce(
    const float* __restrict__ partial, int n, float* __restrict__ gmax) {
  __shared__ float red[256];
  float m = NEG_INF;
  for (int i = threadIdx.x; i < n; i += 256) m = fmaxf(m, partial[i]);
  red[threadIdx.x] = m;
  __syncthreads();
  for (int s = 128; s > 0; s >>= 1) {
    if (threadIdx.x < s)
      red[threadIdx.x] = fmaxf(red[threadIdx.x], red[threadIdx.x + s]);
    __syncthreads();
  }
  if (threadIdx.x == 0) *gmax = red[0];
}

// ---------------------------------------------------------------------------
// Kernel C: ctxT[e,m] += sum_n k'[n,m] v[n,e];  ksum[m] += sum_n k'[n,m].
// grid = BH*8 (8-way N split), block = 256 (8 waves).
// ---------------------------------------------------------------------------
__global__ __launch_bounds__(256) void perf_context(
    const float* __restrict__ k, const float* __restrict__ v,
    const float* __restrict__ proj, const float* __restrict__ gmax_p,
    float* __restrict__ ctxT, float* __restrict__ ksum) {
  __shared__ alignas(16) _Float16 p_lds[DDIM * MFEA];   // B-frag layout, 32KB
  __shared__ alignas(16) _Float16 a_lds[32 * DDIM];     // k rows f16
  __shared__ alignas(16) _Float16 v_lds[32 * DDIM];     // v rows f16 [n][e]
  __shared__ alignas(16) _Float16 kp_lds[MFEA * 32];    // k' transposed [m][n]
  __shared__ float ksum_l[MFEA];
  __shared__ float diag_l[32];

  const int tid = threadIdx.x, lane = tid & 31, w = tid >> 5;
  const int bh = blockIdx.x >> 3, s = blockIdx.x & 7;
  const float gmax = *gmax_p;

  fill_pfrag(p_lds, proj, tid);
  ksum_l[tid] = 0.f;

  // hoisted loop-invariant projection B-fragments for this wave's m-tiles
  const v16h pb00 = ld_bfrag(p_lds + ((2 * w + 0) * 2 + 0) * 512, lane);
  const v16h pb01 = ld_bfrag(p_lds + ((2 * w + 0) * 2 + 1) * 512, lane);
  const v16h pb10 = ld_bfrag(p_lds + ((2 * w + 1) * 2 + 0) * 512, lane);
  const v16h pb11 = ld_bfrag(p_lds + ((2 * w + 1) * 2 + 1) * 512, lane);

  v8f acc2[2][4];
#pragma unroll
  for (int i = 0; i < 2; ++i)
#pragma unroll
    for (int j = 0; j < 4; ++j) acc2[i][j] = zero8();

  const size_t rowbase = ((size_t)bh * NSEQ + (size_t)s * 512) * DDIM;
  const float* kp = k + rowbase;
  const float* vp = v + rowbase;
  __syncthreads();

  for (int step = 0; step < 512; step += 32) {
    {  // load 32 rows of k,v (packed b128 stores) + per-row diag
      const int r = tid >> 3, c0 = (tid & 7) * 8;
      const size_t idx = (size_t)(step + r) * DDIM + c0;
      float ss = 0.f;
      v8h hk, hv;
#pragma unroll
      for (int j = 0; j < 8; ++j) {
        const float kv = kp[idx + j];
        hk[j] = (_Float16)kv;
        hv[j] = (_Float16)vp[idx + j];
        ss += kv * kv;
      }
      *(v8h*)(a_lds + r * DDIM + c0) = hk;
      *(v8h*)(v_lds + r * DDIM + c0) = hv;
      ss += __shfl_down(ss, 4, 32);
      ss += __shfl_down(ss, 2, 32);
      ss += __shfl_down(ss, 1, 32);
      if ((tid & 7) == 0) diag_l[r] = DIAGC * ss;
    }
    __syncthreads();

    // stage 1: k' for m-tiles {2w, 2w+1}; packed b128 transpose-stores
    const v16h a00 = ld_a16(a_lds, DDIM, lane, 0);
    const v16h a01 = ld_a16(a_lds, DDIM, lane, 32);
    const v16h a10 = ld_a16(a_lds + 16 * DDIM, DDIM, lane, 0);
    const v16h a11 = ld_a16(a_lds + 16 * DDIM, DDIM, lane, 32);
#pragma unroll
    for (int mtl = 0; mtl < 2; ++mtl) {
      const int mt = 2 * w + mtl;
      const v16h b0 = mtl ? pb10 : pb00;
      const v16h b1 = mtl ? pb11 : pb01;
      float colsum = 0.f;
#pragma unroll
      for (int ns = 0; ns < 2; ++ns) {
        v8f acc = zero8();
        acc = wmma_f16(ns ? a10 : a00, b0, acc);
        acc = wmma_f16(ns ? a11 : a01, b1, acc);
        const int nrow0 = ns * 16 + ((lane < 16) ? 0 : 8);
        v8h pk;
#pragma unroll
        for (int e = 0; e < 8; ++e) {
          const float val =
              RATIO * (expf(acc[e] - diag_l[nrow0 + e] - gmax) + FEPS);
          pk[e] = (_Float16)val;
          colsum += val;
        }
        // 8 consecutive n for fixed m -> one b128 store
        *(v8h*)(kp_lds + (mt * 16 + (lane & 15)) * 32 + nrow0) = pk;
      }
      colsum += __shfl_xor(colsum, 16, 32);
      if (lane < 16) ksum_l[mt * 16 + lane] += colsum;
    }
    __syncthreads();

    // stage 2: acc2 += k'ᵀ[m-tile, 32n] x v[32n, e-tile]
    const v16h bv0 = ld_b16s(v_lds + 0,  DDIM, lane, 0);
    const v16h bv1 = ld_b16s(v_lds + 16, DDIM, lane, 0);
    const v16h bv2 = ld_b16s(v_lds + 32, DDIM, lane, 0);
    const v16h bv3 = ld_b16s(v_lds + 48, DDIM, lane, 0);
#pragma unroll
    for (int mtl = 0; mtl < 2; ++mtl) {
      const v16h av = ld_a16(kp_lds + (2 * w + mtl) * 16 * 32, 32, lane, 0);
      acc2[mtl][0] = wmma_f16(av, bv0, acc2[mtl][0]);
      acc2[mtl][1] = wmma_f16(av, bv1, acc2[mtl][1]);
      acc2[mtl][2] = wmma_f16(av, bv2, acc2[mtl][2]);
      acc2[mtl][3] = wmma_f16(av, bv3, acc2[mtl][3]);
    }
    __syncthreads();
  }

  // flush accumulators into transposed context [e][m]
#pragma unroll
  for (int mtl = 0; mtl < 2; ++mtl)
#pragma unroll
    for (int et = 0; et < 4; ++et)
#pragma unroll
      for (int e = 0; e < 8; ++e) {
        const int m   = (2 * w + mtl) * 16 + ((lane < 16) ? e : (e + 8));
        const int col = et * 16 + (lane & 15);
        atomicAdd(&ctxT[((size_t)bh * EDIM + col) * MFEA + m],
                  acc2[mtl][et][e]);
      }
  atomicAdd(&ksum[bh * MFEA + tid], ksum_l[tid]);
}

// ---------------------------------------------------------------------------
// Kernel D: q' + out = Dinv * q'·ctx, with GEMM1 computed transposed
// (P-as-A x q-as-B -> [m][n] tiles): per-lane scalar row stats and fully
// vectorized q' stores. grid = BH*32 (128 q rows), dynamic LDS 149504 B.
// ---------------------------------------------------------------------------
__global__ __launch_bounds__(256) void perf_out(
    const float* __restrict__ q, const float* __restrict__ proj,
    const float* __restrict__ ctxT, const float* __restrict__ ksum,
    float* __restrict__ out) {
  extern __shared__ char smem[];
  _Float16* p_lds  = (_Float16*)(smem);            // 32768  proj [m][d] scaled
  _Float16* qf_lds = (_Float16*)(smem + 32768);    // 16384  q B-frag layout
  _Float16* qp_lds = (_Float16*)(smem + 49152);    // 65536  q' [n][m]
  _Float16* ctx_l  = (_Float16*)(smem + 114688);   // 32768  ctxT [e][m]
  float*    ksum_l = (float*)(smem + 147456);      // 1024
  float*    diag_l = (float*)(smem + 148480);      // 512
  float*    dinv_l = (float*)(smem + 148992);      // 512

  const int tid = threadIdx.x, lane = tid & 31, w = tid >> 5;
  const int bh = blockIdx.x >> 5, blk = blockIdx.x & 31;
  const int row0 = blk * 128;

  {  // projection rows (natural A layout), scaled, packed stores
    const float* src = proj + tid * DDIM;
    _Float16* dst = p_lds + tid * DDIM;
#pragma unroll
    for (int g = 0; g < 8; ++g) {
      v8h h;
#pragma unroll
      for (int j = 0; j < 8; ++j)
        h[j] = (_Float16)(src[g * 8 + j] * NORMALIZER);
      *(v8h*)(dst + g * 8) = h;
    }
  }
  {  // ctxT rows [e][m], packed stores
    const int e = tid >> 2, m0 = (tid & 3) * 64;
    const float* src = ctxT + ((size_t)bh * EDIM + e) * MFEA + m0;
    _Float16* dst = ctx_l + e * MFEA + m0;
#pragma unroll
    for (int g = 0; g < 8; ++g) {
      v8h h;
#pragma unroll
      for (int j = 0; j < 8; ++j) h[j] = (_Float16)src[g * 8 + j];
      *(v8h*)(dst + g * 8) = h;
    }
  }
  ksum_l[tid] = ksum[bh * MFEA + tid];
  {  // q rows into B-fragment layout + diag
    const int r = tid >> 1, c0 = (tid & 1) * 32;
    const float* qrow = q + ((size_t)bh * NSEQ + row0 + r) * DDIM;
    float x[32];
    float ss = 0.f;
#pragma unroll
    for (int j = 0; j < 32; ++j) {
      x[j] = qrow[c0 + j];
      ss += x[j] * x[j];
    }
#pragma unroll
    for (int sgrp = 0; sgrp < 2; ++sgrp) {   // two 16-d subgroups
      const int d0 = c0 + sgrp * 16;
      const int tile  = (r >> 4) * 2 + (d0 >> 5);
      const int lslot = (r & 15) + ((d0 & 16) ? 16 : 0);
      _Float16* dst = qf_lds + (tile * 32 + lslot) * 16;
      v8h h0, h1;
#pragma unroll
      for (int j = 0; j < 8; ++j) {
        h0[j] = (_Float16)x[sgrp * 16 + j];
        h1[j] = (_Float16)x[sgrp * 16 + 8 + j];
      }
      *(v8h*)dst = h0;
      *((v8h*)dst + 1) = h1;
    }
    ss += __shfl_xor(ss, 1, 32);
    if ((tid & 1) == 0) diag_l[r] = DIAGC * ss;
  }
  __syncthreads();

  // wave w owns query rows w*16..+15 (transposed tiles: rows=m, cols=n)
  const v16h qb0 = ld_bfrag(qf_lds + (w * 2 + 0) * 512, lane);
  const v16h qb1 = ld_bfrag(qf_lds + (w * 2 + 1) * 512, lane);

  // pass 1: per-query max over m (per-lane scalar + one half-swap)
  float rmaxn = NEG_INF;
  for (int mt = 0; mt < 16; ++mt) {
    v8f acc = zero8();
    acc = wmma_f16(ld_a16(p_lds + mt * 16 * DDIM, DDIM, lane, 0),  qb0, acc);
    acc = wmma_f16(ld_a16(p_lds + mt * 16 * DDIM, DDIM, lane, 32), qb1, acc);
#pragma unroll
    for (int e = 0; e < 8; ++e) rmaxn = fmaxf(rmaxn, acc[e]);
  }
  rmaxn = fmaxf(rmaxn, __shfl_xor(rmaxn, 16, 32));
  const float dq = diag_l[w * 16 + (lane & 15)];

  // pass 2: recompute, exp, packed b128 q' stores (8 consecutive m per lane)
  for (int mt = 0; mt < 16; ++mt) {
    v8f acc = zero8();
    acc = wmma_f16(ld_a16(p_lds + mt * 16 * DDIM, DDIM, lane, 0),  qb0, acc);
    acc = wmma_f16(ld_a16(p_lds + mt * 16 * DDIM, DDIM, lane, 32), qb1, acc);
    v8h pk;
#pragma unroll
    for (int e = 0; e < 8; ++e)
      pk[e] = (_Float16)(RATIO * (expf(acc[e] - dq - rmaxn) + FEPS));
    *(v8h*)(qp_lds + (w * 16 + (lane & 15)) * MFEA + mt * 16 +
            ((lane < 16) ? 0 : 8)) = pk;
  }
  __syncthreads();

  // denominator: Dinv[n] = 1 / (q'[n,:] . ksum)
  {
    const int r = tid >> 1, m0 = (tid & 1) * 128;
    float den = 0.f;
    for (int j = 0; j < 128; ++j)
      den += (float)qp_lds[r * MFEA + m0 + j] * ksum_l[m0 + j];
    den += __shfl_xor(den, 1, 32);
    if ((tid & 1) == 0) dinv_l[r] = 1.0f / den;
  }
  __syncthreads();

  // GEMM3: out-tile = q'[16 x 256] x ctx[256 x 64] (B from transposed ctx)
  v8f oacc[4];
#pragma unroll
  for (int et = 0; et < 4; ++et) oacc[et] = zero8();
  for (int kb = 0; kb < MFEA; kb += 32) {
    const v16h aq = ld_a16(qp_lds + w * 16 * MFEA, MFEA, lane, kb);
#pragma unroll
    for (int et = 0; et < 4; ++et) {
      const v16h bc = ld_bT(ctx_l + et * 16 * MFEA, MFEA, lane, kb);
      oacc[et] = wmma_f16(aq, bc, oacc[et]);
    }
  }
  float* orow = out + ((size_t)bh * NSEQ + row0 + w * 16) * EDIM;
#pragma unroll
  for (int et = 0; et < 4; ++et)
#pragma unroll
    for (int e = 0; e < 8; ++e) {
      const int r = (lane < 16) ? e : (e + 8);
      orow[(size_t)r * EDIM + et * 16 + (lane & 15)] =
          oacc[et][e] * dinv_l[w * 16 + r];
    }
}

// ---------------------------------------------------------------------------
extern "C" void kernel_launch(void* const* d_in, const int* in_sizes, int n_in,
                              void* d_out, int out_size, void* d_ws,
                              size_t ws_size, hipStream_t stream) {
  const float* q    = (const float*)d_in[0];
  const float* k    = (const float*)d_in[1];
  const float* v    = (const float*)d_in[2];
  const float* proj = (const float*)d_in[3];
  float* out = (float*)d_out;

  // workspace layout (floats): ctxT | ksum | partials | gmax  (~4.1 MB)
  float* ctxT    = (float*)d_ws;                     // BH*64*256
  float* ksum    = ctxT + (size_t)BH * EDIM * MFEA;  // BH*256
  float* partial = ksum + (size_t)BH * MFEA;         // BH*32
  float* gmax    = partial + BH * 32;                // 1

  const int zero_n = BH * EDIM * MFEA + BH * MFEA;   // ctxT+ksum contiguous
  perf_zero<<<dim3(4096), dim3(256), 0, stream>>>(ctxT, zero_n);

  perf_kmax<<<dim3(BH * 32), dim3(256), 0, stream>>>(k, proj, partial);
  perf_maxreduce<<<dim3(1), dim3(256), 0, stream>>>(partial, BH * 32, gmax);
  perf_context<<<dim3(BH * 8), dim3(256), 0, stream>>>(k, v, proj, gmax,
                                                       ctxT, ksum);
  const size_t smemD = 149504;  // dynamic LDS (<=320KB/WGP on gfx1250)
  perf_out<<<dim3(BH * 32), dim3(256), smemD, stream>>>(q, proj, ctxT, ksum,
                                                        out);
}
